// ncm_output_11914239279125
// MI455X (gfx1250) — matrix operand order repaired and checked by reading, not compiled, not run on hardware
//
#include <hip/hip_runtime.h>
#include <cstdint>
#include <cstddef>

// ---------------------------------------------------------------- types
typedef __attribute__((ext_vector_type(16))) __bf16 v16bf;
typedef __attribute__((ext_vector_type(4)))  __bf16 v4bf;
typedef __attribute__((ext_vector_type(8)))  float  v8f;
typedef __attribute__((ext_vector_type(4)))  float  v4f;
typedef __attribute__((ext_vector_type(4)))  unsigned int u32x4;
typedef __attribute__((ext_vector_type(4)))  int i32x4;

constexpr int NB = 8192;   // batch rows
constexpr int ND = 1024;   // feature dim (K)
constexpr int NC = 4096;   // centroids (output cols)

constexpr int TM = 128;    // block tile M
constexpr int TN = 128;    // block tile N
constexpr int TK = 32;     // K step (bf16 WMMA K)

// ---------------------------------------------------------------- async-LDS path selection
#if defined(__AMDGCN__) && __has_builtin(__builtin_amdgcn_global_load_async_to_lds_b128)
#define ASYNC_LDS 1
#pragma message("gfx1250: using global_load_async_to_lds_b128 staging path")
#else
#define ASYNC_LDS 0
#pragma message("gfx1250: async-to-LDS builtin unavailable; using load+ds_store fallback")
#endif

__device__ __forceinline__ void wait_async_lds() {
#if ASYNC_LDS
#if __has_builtin(__builtin_amdgcn_s_wait_asynccnt)
  __builtin_amdgcn_s_wait_asynccnt(0);
#else
  asm volatile("s_wait_asynccnt 0" ::: "memory");
#endif
#endif
}

__device__ __forceinline__ void stage16(const __bf16* gsrc, __bf16* ldst) {
#if ASYNC_LDS
  typedef __attribute__((address_space(1))) i32x4 GI;   // global int4
  typedef __attribute__((address_space(3))) i32x4 LI;   // LDS int4
  __builtin_amdgcn_global_load_async_to_lds_b128(
      (GI*)(uintptr_t)gsrc,
      (LI*)(unsigned int)(uintptr_t)ldst,   // generic LDS ptr low 32 bits == LDS offset
      0, 0);
#else
  *(u32x4*)ldst = *(const u32x4*)gsrc;
#endif
}

// ---------------------------------------------------------------- kernel 1: x -> bf16, x2 = rowwise ||x||^2
__global__ __launch_bounds__(256) void k_prep_x(const float* __restrict__ x,
                                                __bf16* __restrict__ xb,
                                                float* __restrict__ x2) {
  __shared__ float red[256];
  const int r = blockIdx.x;
  const int t = threadIdx.x;
  const v4f* px = (const v4f*)(x + (size_t)r * ND);
  v4f v = px[t];                       // 256 threads * 4 = 1024 elements
  v4bf h = { (__bf16)v.x, (__bf16)v.y, (__bf16)v.z, (__bf16)v.w };
  ((v4bf*)(xb + (size_t)r * ND))[t] = h;
  red[t] = v.x*v.x + v.y*v.y + v.z*v.z + v.w*v.w;
  __syncthreads();
  for (int o = 128; o > 0; o >>= 1) {
    if (t < o) red[t] += red[t + o];
    __syncthreads();
  }
  if (t == 0) x2[r] = red[0];
}

// ---------------------------------------------------------------- kernel 2: cent (bf16) via 32x32 LDS transpose
// cent[i,j] = weight[(i%4)*1024 + j, i/4]   (closed form of weight.T.reshape(C,D))
// For a weight tile (r0..r0+31, c0..c0+31): element (r,c) -> cent(4c + (r>>10), r&1023)
__global__ __launch_bounds__(256) void k_cent(const float* __restrict__ w,
                                              __bf16* __restrict__ cb) {
  __shared__ float tile[32][33];
  const int tr = blockIdx.x & 127;  // weight row-tile  (4096/32)
  const int tc = blockIdx.x >> 7;   // weight col-tile  (1024/32)
  const int tx = threadIdx.x & 31;
  const int ty = threadIdx.x >> 5;  // 0..7
#pragma unroll
  for (int it = 0; it < 4; ++it) {
    int rr = ty + it * 8;
    tile[rr][tx] = w[(size_t)(tr * 32 + rr) * ND + tc * 32 + tx]; // coalesced read
  }
  __syncthreads();
  const int q  = (tr * 32) >> 10;
  const int j0 = (tr * 32) & 1023;
#pragma unroll
  for (int it = 0; it < 4; ++it) {
    int cc = ty + it * 8;
    int i = 4 * (tc * 32 + cc) + q;
    cb[(size_t)i * ND + j0 + tx] = (__bf16)tile[tx][cc];          // coalesced write
  }
}

// ---------------------------------------------------------------- kernel 3: c2 = rowwise ||cent||^2
__global__ __launch_bounds__(256) void k_c2(const __bf16* __restrict__ cb,
                                            float* __restrict__ c2) {
  __shared__ float red[256];
  const int r = blockIdx.x;
  const int t = threadIdx.x;
  v4bf h = ((const v4bf*)(cb + (size_t)r * ND))[t];
  float a = (float)h.x, b = (float)h.y, c = (float)h.z, d = (float)h.w;
  red[t] = a*a + b*b + c*c + d*d;
  __syncthreads();
  for (int o = 128; o > 0; o >>= 1) {
    if (t < o) red[t] += red[t + o];
    __syncthreads();
  }
  if (t == 0) c2[r] = red[0];
}

// ---------------------------------------------------------------- kernel 4: GEMM + fused epilogue
// out[b,c] = 2*xc - x2[b] - c2[c] - bias[c]
__device__ __forceinline__ void stage_tile(const __bf16* __restrict__ gA,
                                           const __bf16* __restrict__ gB,
                                           __bf16* sA, __bf16* sB, int tid) {
  // A tile: 128 rows x 32 bf16 = 512 chunks of 16B; same for B. 2 chunks/thread each.
#pragma unroll
  for (int c = tid; c < 512; c += 256) {
    int row = c >> 2;
    int k8  = (c & 3) << 3;
    stage16(gA + (size_t)row * ND + k8, sA + row * TK + k8);
    stage16(gB + (size_t)row * ND + k8, sB + row * TK + k8);
  }
}

__global__ __launch_bounds__(256) void k_gemm(const __bf16* __restrict__ xb,
                                              const __bf16* __restrict__ cb,
                                              const float* __restrict__ x2,
                                              const float* __restrict__ c2,
                                              const float* __restrict__ bias,
                                              float* __restrict__ out) {
  __shared__ __bf16 sA[2][TM * TK];   // 2 x 8 KB
  __shared__ __bf16 sB[2][TN * TK];   // 2 x 8 KB
  const int tid  = threadIdx.x;
  const int bm   = (blockIdx.x & 63) * TM;   // 8192/128 = 64 M tiles
  const int bn   = (blockIdx.x >> 6) * TN;   // 4096/128 = 32 N tiles
  const int w    = tid >> 5;                 // wave id 0..7 (wave32)
  const int lane = tid & 31;
  const int wm   = (w & 1) * 64;             // 2 waves along M, 64 rows each
  const int wn   = (w >> 1) * 32;            // 4 waves along N, 32 cols each
  const int half = lane >> 4;                // K half-select per ISA 16-bit layouts
  const int l16  = lane & 15;

  const __bf16* gA = xb + (size_t)bm * ND;
  const __bf16* gB = cb + (size_t)bn * ND;

  v8f acc[4][2] = {};

  stage_tile(gA, gB, sA[0], sB[0], tid);
  wait_async_lds();
  __syncthreads();

  for (int kk = 0; kk < ND / TK; ++kk) {
    const int cur = kk & 1;
    if (kk + 1 < ND / TK)
      stage_tile(gA + (kk + 1) * TK, gB + (kk + 1) * TK,
                 sA[cur ^ 1], sB[cur ^ 1], tid);

    // A fragment (16x32 bf16): lanes 0-15 -> K[0..7],K[16..23]; lanes 16-31 -> K[8..15],K[24..31]
    v16bf va[4], vb[2];
#pragma unroll
    for (int mt = 0; mt < 4; ++mt) {
      const __bf16* p = &sA[cur][(wm + mt * 16 + l16) * TK + half * 8];
      union { u32x4 q[2]; v16bf v; } u;
      u.q[0] = *(const u32x4*)p;          // 8 bf16
      u.q[1] = *(const u32x4*)(p + 16);   // +32B -> K+16 block
      va[mt] = u.v;
    }
    // B fragment (32x16 bf16): lanes 0-15 -> K[0..7],K[16..23]; lanes 16-31 -> K[8..15],K[24..31]
#pragma unroll
    for (int nt = 0; nt < 2; ++nt) {
      const __bf16* p = &sB[cur][(wn + nt * 16 + l16) * TK + half * 8];
      union { u32x4 q[2]; v16bf v; } u;
      u.q[0] = *(const u32x4*)p;
      u.q[1] = *(const u32x4*)(p + 16);   // +32B -> K+16 block
      vb[nt] = u.v;
    }
#pragma unroll
    for (int mt = 0; mt < 4; ++mt)
#pragma unroll
      for (int nt = 0; nt < 2; ++nt)
        acc[mt][nt] = __builtin_amdgcn_wmma_f32_16x16x32_bf16(
            false, va[mt], false, vb[nt], (short)0, acc[mt][nt], false, false);

    wait_async_lds();
    __syncthreads();
  }

  // Epilogue: D layout = VGPR v: M = v + half*8, N = l16
#pragma unroll
  for (int nt = 0; nt < 2; ++nt) {
    const int col = bn + wn + nt * 16 + l16;
    const float cbv = c2[col] + bias[col];
#pragma unroll
    for (int mt = 0; mt < 4; ++mt) {
      const int rbase = bm + wm + mt * 16 + half * 8;
      float* po = out + (size_t)rbase * NC + col;
#pragma unroll
      for (int v = 0; v < 8; ++v)
        po[(size_t)v * NC] = 2.0f * acc[mt][nt][v] - x2[rbase + v] - cbv;
    }
  }
}

// ---------------------------------------------------------------- launch
extern "C" void kernel_launch(void* const* d_in, const int* in_sizes, int n_in,
                              void* d_out, int out_size, void* d_ws, size_t ws_size,
                              hipStream_t stream) {
  const float* x    = (const float*)d_in[0];   // (8192, 1024) f32
  const float* wgt  = (const float*)d_in[1];   // (4096, 1024) f32
  const float* bias = (const float*)d_in[2];   // (4096,) f32
  float* out = (float*)d_out;                  // (8192, 4096) f32

  char* ws = (char*)d_ws;
  __bf16* xb  = (__bf16*)ws;                                        // 16 MB
  __bf16* cbf = (__bf16*)(ws + (size_t)NB * ND * 2);                // 8 MB
  float*  x2  = (float*)(ws + (size_t)NB * ND * 2 + (size_t)NC * ND * 2); // 32 KB
  float*  c2  = x2 + NB;                                            // 16 KB

  hipLaunchKernelGGL(k_prep_x, dim3(NB), dim3(256), 0, stream, x, xb, x2);
  hipLaunchKernelGGL(k_cent, dim3((NC / 32) * (ND / 32)), dim3(256), 0, stream, wgt, cbf);
  hipLaunchKernelGGL(k_c2, dim3(NC), dim3(256), 0, stream, cbf, c2);
  hipLaunchKernelGGL(k_gemm, dim3((NB / TM) * (NC / TN)), dim3(256), 0, stream,
                     xb, cbf, x2, c2, bias, out);
}
